// MultiHeadAttention_64759516889590
// MI455X (gfx1250) — matrix operand order repaired and checked by reading, not compiled, not run on hardware
//
#include <hip/hip_runtime.h>
#include <stdint.h>

// ---------------------------------------------------------------------------
// MultiHeadAttention forward for MI455X (gfx1250), bf16 WMMA pipeline.
//   0) f32 -> bf16 converts (q,k,v) and transpose-convert weights
//   1) QKV projections (bf16 GEMM, head-major outputs; V stored transposed)
//   2) scores = Q·K^T + mask -> f32 into atten region of d_out
//   3) row softmax in place
//   4) ctx = atten·V (A converted f32->bf16 during LDS staging)
//   5) out-proj GEMM -> f32 scratch
//   6) residual add + LayerNorm -> d_out
// bf16 GEMMs: double-buffered LDS tiles fed by the Tensor Data Mover; DMA of
// tile i+1 overlaps WMMA on tile i (s_wait_tensorcnt <= 1, in-order per wave).
// ---------------------------------------------------------------------------

typedef __bf16 bf16;
typedef float  v8f   __attribute__((ext_vector_type(8)));
typedef __bf16 v16bf __attribute__((ext_vector_type(16)));

constexpr int BB = 4;          // batch
constexpr int LL = 2048;       // seq len
constexpr int CC = 512;        // model dim
constexpr int HH = 8;          // heads
constexpr int DD = 512;        // head dim
constexpr int HDc = HH * DD;   // 4096
constexpr int BLc = BB * LL;   // 8192
constexpr float EPSc = 1e-5f;

constexpr int BK   = 32;       // K-slice per stage (one WMMA k-step)
constexpr int LDSS = BK + 8;   // 80B row stride: 16B aligned, conflict-free

// --------------------------- TDM (Tensor Data Mover) -----------------------
#if __has_builtin(__builtin_amdgcn_tensor_load_to_lds) && \
    __has_builtin(__builtin_amdgcn_s_wait_tensorcnt)
#define ATHENA_TDM 1
typedef unsigned int u32x4_t __attribute__((ext_vector_type(4)));
typedef int          i32x8_t __attribute__((ext_vector_type(8)));
typedef int          i32x4_t __attribute__((ext_vector_type(4)));

// DMA a 128-row x 32-element bf16 tile (row stride = stride_elems in memory)
// into LDS at byte offset lds_addr, padding the LDS destination by 4 DWORDs
// (8 bf16) after every 16 DWORDs (one 32-elem row) -> 40-element LDS rows.
// D# layout per CDNA5 ISA ch.8: group0 = {flags, lds_addr, global_addr, type},
// group1 = {mask/data_size/pad, tensor dims, tile dims, strides}.
// This toolchain's builtin takes 6 args (groups 0..3 + extra dwords + cpol);
// trailing operands are zero (2D tile, no gather/iterate/multicast).
__device__ __forceinline__ void tdm_load_tile(uint32_t lds_addr, uint64_t gaddr,
                                              uint32_t stride_elems) {
  u32x4_t g0;
  g0[0] = 1u;                                        // count=1 (valid user D#)
  g0[1] = lds_addr;                                  // lds_addr [63:32]
  g0[2] = (uint32_t)gaddr;                           // global_addr[31:0]
  g0[3] = (uint32_t)((gaddr >> 32) & 0x01FFFFFFu)    // global_addr[56:32]
          | 0x80000000u;                             // type=2 ("image")
  i32x8_t g1;
  g1[0] = (int)((1u << 16)      // data_size=1 -> 2-byte elements
                | (1u << 20)    // pad_enable
                | (3u << 22)    // pad_interval: code 3 -> every 16 DWORDs
                | (3u << 25));  // pad_amount:   code 3 -> 4 DWORDs (16B)
  g1[1] = (int)0xFFFF0000u;                   // tensor_dim0 = 0x7FFFFFFF (lo16)
  g1[2] = (int)(0x7FFFu | 0xFFFF0000u);       // tensor_dim0 hi | tensor_dim1 lo
  g1[3] = (int)(0x7FFFu | (32u << 16));       // tensor_dim1 hi | tile_dim0=32
  g1[4] = (int)128u;                          // tile_dim1=128, tile_dim2=0 (2D)
  g1[5] = (int)stride_elems;                  // tensor_dim0_stride[31:0]
  g1[6] = 0;                                  // stride[47:32] | dim1_stride lo
  g1[7] = 0;
  i32x4_t gz4 = {0, 0, 0, 0};                 // groups 2/3 unused for 2D
  i32x8_t gz8 = {0, 0, 0, 0, 0, 0, 0, 0};     // extra operand (zero-filled)
  __builtin_amdgcn_tensor_load_to_lds(g0, g1, gz4, gz4, gz8, 0);
}
#endif

// ------------------------------ fragment helpers ---------------------------
// ISA 16-bit A 16x32 layout: lane<16 -> row=lane, K 0..7 (VGPR0-3) and 16..23
// (VGPR4-7); lane>=16 -> row=lane-16, K 8..15 and 24..31.
// B 32x16 (stored as B^T rows of 32 K-elems): lane<16 -> col=lane, K 0..7 and
// 16..23; lane>=16 -> col=lane-16, K 8..15 and 24..31 (same order as A).
struct FragBits { uint4 lo, hi; };

__device__ __forceinline__ v16bf frag_a(const bf16* row, int hl) {
  FragBits f;
  f.lo = *reinterpret_cast<const uint4*>(row + hl * 8);
  f.hi = *reinterpret_cast<const uint4*>(row + 16 + hl * 8);
  return __builtin_bit_cast(v16bf, f);
}

__device__ __forceinline__ v16bf frag_b(const bf16* row, int hl) {
  FragBits f;
  f.lo = *reinterpret_cast<const uint4*>(row + hl * 8);
  f.hi = *reinterpret_cast<const uint4*>(row + 16 + hl * 8);
  return __builtin_bit_cast(v16bf, f);
}

__device__ __forceinline__ v8f wmma_bf16(v16bf a, v16bf b, v8f c) {
  return __builtin_amdgcn_wmma_f32_16x16x32_bf16(
      /*neg_a=*/false, a, /*neg_b=*/false, b,
      /*c_mod=*/(short)0, c, /*reuse_a=*/false, /*reuse_b=*/false);
}

// One 32-K step over a 128x128 block tile: wave (wm,wn) does 4x2 WMMAs.
__device__ __forceinline__ void compute_step(const bf16 (*As)[LDSS],
                                             const bf16 (*Bs)[LDSS],
                                             int wm, int wn, int r, int hl,
                                             v8f acc[4][2]) {
  v16bf af[4], bfr[2];
#pragma unroll
  for (int i = 0; i < 4; ++i)
    af[i] = frag_a(&As[wm * 64 + i * 16 + r][0], hl);
#pragma unroll
  for (int j = 0; j < 2; ++j)
    bfr[j] = frag_b(&Bs[wn * 32 + j * 16 + r][0], hl);
#pragma unroll
  for (int i = 0; i < 4; ++i)
#pragma unroll
    for (int j = 0; j < 2; ++j)
      acc[i][j] = wmma_bf16(af[i], bfr[j], acc[i][j]);
}

// ------------------------------- GEMM core ---------------------------------
// Block tile 128x128, 256 threads = 8 waves (wave32); wave grid 2x4, each
// wave owns 64x32 -> 4x2 wmma tiles (v8f accumulators).
// A: row-major [*, lda] (bf16, or f32 converted during staging when AF32).
// BT: row-major [N, ldb] = B transposed, so K is contiguous.
template <bool AF32>
__device__ __forceinline__ void gemm_core(const void* Ag, int lda,
                                          const bf16* BTg, int ldb, int K,
                                          int m0, int n0, v8f acc[4][2]) {
  const int tid = threadIdx.x;
  const int lane = tid & 31;
  const int hl = lane >> 4;
  const int r = lane & 15;
  // scalar (SGPR) wave id -> scalar branches around TDM issue
  const int wid = __builtin_amdgcn_readfirstlane((unsigned)tid) >> 5;
  const int wave = tid >> 5;
  const int wm = wave >> 2;         // 0..1
  const int wn = wave & 3;          // 0..3

#if defined(ATHENA_TDM)
  if constexpr (!AF32) {
    // ---- software-pipelined: TDM double buffer, DMA(i+1) overlaps WMMA(i) --
    __shared__ alignas(16) bf16 As[2][128][LDSS];
    __shared__ alignas(16) bf16 Bs[2][128][LDSS];
    const uint32_t ldsA0 = (uint32_t)(uintptr_t)(&As[0][0][0]);
    const uint32_t ldsA1 = (uint32_t)(uintptr_t)(&As[1][0][0]);
    const uint32_t ldsB0 = (uint32_t)(uintptr_t)(&Bs[0][0][0]);
    const uint32_t ldsB1 = (uint32_t)(uintptr_t)(&Bs[1][0][0]);
    const uint64_t gA = (uint64_t)(uintptr_t)Ag + (size_t)m0 * lda * 2;
    const uint64_t gB = (uint64_t)(uintptr_t)BTg + (size_t)n0 * ldb * 2;

    if (wid == 0)      tdm_load_tile(ldsA0, gA, (uint32_t)lda);
    else if (wid == 1) tdm_load_tile(ldsB0, gB, (uint32_t)ldb);

    int cur = 0;
    for (int k0 = 0; k0 < K - BK; k0 += BK) {
      // issue next tile into the other buffer, then wait for current tile:
      // per-wave TDM ops complete in order, so TENSORcnt<=1 => tile(i) done.
      if (wid == 0) {
        tdm_load_tile(cur ? ldsA0 : ldsA1, gA + (size_t)(k0 + BK) * 2,
                      (uint32_t)lda);
        __builtin_amdgcn_s_wait_tensorcnt(1);
      } else if (wid == 1) {
        tdm_load_tile(cur ? ldsB0 : ldsB1, gB + (size_t)(k0 + BK) * 2,
                      (uint32_t)ldb);
        __builtin_amdgcn_s_wait_tensorcnt(1);
      }
      __syncthreads();
      compute_step(As[cur], Bs[cur], wm, wn, r, hl, acc);
      cur ^= 1;
      __syncthreads();
    }
    if (wid <= 1) __builtin_amdgcn_s_wait_tensorcnt(0);
    __syncthreads();
    compute_step(As[cur], Bs[cur], wm, wn, r, hl, acc);
    return;
  } else {
    // ---- A: f32 -> bf16 VALU staging; B: TDM (single buffer) ----
    __shared__ alignas(16) bf16 As[128][LDSS];
    __shared__ alignas(16) bf16 Bs[128][LDSS];
    const uint32_t ldsB = (uint32_t)(uintptr_t)(&Bs[0][0]);
    const uint64_t gB = (uint64_t)(uintptr_t)BTg + (size_t)n0 * ldb * 2;
    const int sr = tid >> 1;          // staging row 0..127
    const int sc = (tid & 1) * 16;    // staging col 0 or 16 (16 elems each)

    for (int k0 = 0; k0 < K; k0 += BK) {
      const float* src = (const float*)Ag + (size_t)(m0 + sr) * lda + k0 + sc;
      float4 f[4];
#pragma unroll
      for (int u = 0; u < 4; ++u)
        f[u] = *reinterpret_cast<const float4*>(src + u * 4);
#pragma unroll
      for (int u = 0; u < 4; u += 2) {
        union { bf16 h[8]; uint4 q; } cv;
        cv.h[0] = (bf16)f[u].x;     cv.h[1] = (bf16)f[u].y;
        cv.h[2] = (bf16)f[u].z;     cv.h[3] = (bf16)f[u].w;
        cv.h[4] = (bf16)f[u + 1].x; cv.h[5] = (bf16)f[u + 1].y;
        cv.h[6] = (bf16)f[u + 1].z; cv.h[7] = (bf16)f[u + 1].w;
        *reinterpret_cast<uint4*>(&As[sr][sc + u * 4]) = cv.q;
      }
      if (wid == 0) {
        tdm_load_tile(ldsB, gB + (size_t)k0 * 2, (uint32_t)ldb);
        __builtin_amdgcn_s_wait_tensorcnt(0);
      }
      __syncthreads();
      compute_step(As, Bs, wm, wn, r, hl, acc);
      __syncthreads();
    }
    return;
  }
#else
  // ---------------- fallback: VGPR staging, single buffer ----------------
  __shared__ alignas(16) bf16 As[128][LDSS];
  __shared__ alignas(16) bf16 Bs[128][LDSS];
  const int sr = tid >> 1;
  const int sc = (tid & 1) * 16;
  for (int k0 = 0; k0 < K; k0 += BK) {
    if constexpr (AF32) {
      const float* src = (const float*)Ag + (size_t)(m0 + sr) * lda + k0 + sc;
      float4 f[4];
#pragma unroll
      for (int u = 0; u < 4; ++u)
        f[u] = *reinterpret_cast<const float4*>(src + u * 4);
#pragma unroll
      for (int u = 0; u < 4; u += 2) {
        union { bf16 h[8]; uint4 q; } cv;
        cv.h[0] = (bf16)f[u].x;     cv.h[1] = (bf16)f[u].y;
        cv.h[2] = (bf16)f[u].z;     cv.h[3] = (bf16)f[u].w;
        cv.h[4] = (bf16)f[u + 1].x; cv.h[5] = (bf16)f[u + 1].y;
        cv.h[6] = (bf16)f[u + 1].z; cv.h[7] = (bf16)f[u + 1].w;
        *reinterpret_cast<uint4*>(&As[sr][sc + u * 4]) = cv.q;
      }
    } else {
      const bf16* src = (const bf16*)Ag + (size_t)(m0 + sr) * lda + k0 + sc;
      uint4 t0 = *reinterpret_cast<const uint4*>(src);
      uint4 t1 = *reinterpret_cast<const uint4*>(src + 8);
      *reinterpret_cast<uint4*>(&As[sr][sc]) = t0;
      *reinterpret_cast<uint4*>(&As[sr][sc + 8]) = t1;
    }
    {
      const bf16* src = BTg + (size_t)(n0 + sr) * ldb + k0 + sc;
      uint4 t0 = *reinterpret_cast<const uint4*>(src);
      uint4 t1 = *reinterpret_cast<const uint4*>(src + 8);
      *reinterpret_cast<uint4*>(&Bs[sr][sc]) = t0;
      *reinterpret_cast<uint4*>(&Bs[sr][sc + 8]) = t1;
    }
    __syncthreads();
    compute_step(As, Bs, wm, wn, r, hl, acc);
    __syncthreads();
  }
#endif
}

// C/D layout: VGPR e, lanes 0-15 -> M=e, N=lane; lanes 16-31 -> M=e+8, N=lane-16.
#define EPI_INDICES                                    \
  const int lane = threadIdx.x & 31;                   \
  const int hl = lane >> 4;                            \
  const int r = lane & 15;                             \
  const int wave = threadIdx.x >> 5;                   \
  const int wm = wave >> 2;                            \
  const int wn = wave & 3;

// ------------------------- conversion kernels ------------------------------
__global__ __launch_bounds__(256) void k_cvt_bf16(const float* s, bf16* d, int n) {
  int i = blockIdx.x * 256 + threadIdx.x;
  if (i < n) d[i] = (bf16)s[i];
}

// s[R][Cc] f32 -> d[Cc][R] bf16
__global__ __launch_bounds__(256) void k_transpose_cvt(const float* s, bf16* d,
                                                       int R, int Cc) {
  int i = blockIdx.x * 256 + threadIdx.x;
  if (i < R * Cc) {
    int rr = i / Cc, cc = i - rr * Cc;
    d[(size_t)cc * R + rr] = (bf16)s[i];
  }
}

// ----------------------------- QKV projection ------------------------------
// A [BL, C] bf16, BT = W^T [HD, C] bf16. Output head-major:
//   transposed==0: out[((b*H+h)*L + l)*D + d]   (for Q, K)
//   transposed==1: out[((b*H+h)*D + d)*L + l]   (for V -> vhT)
__global__ __launch_bounds__(256) void k_gemm_proj(const bf16* A, const bf16* BT,
                                                   const float* bias, bf16* out,
                                                   int transposed) {
  const int m0 = blockIdx.y * 128, n0 = blockIdx.x * 128;
  v8f acc[4][2] = {};
  gemm_core<false>(A, CC, BT, CC, CC, m0, n0, acc);
  EPI_INDICES
#pragma unroll
  for (int i = 0; i < 4; ++i)
#pragma unroll
    for (int j = 0; j < 2; ++j)
#pragma unroll
      for (int e = 0; e < 8; ++e) {
        const int gm = m0 + wm * 64 + i * 16 + hl * 8 + e;
        const int gn = n0 + wn * 32 + j * 16 + r;
        const float val = acc[i][j][e] + bias[gn];
        const int b = gm >> 11, l = gm & (LL - 1);
        const int h = gn >> 9, d = gn & (DD - 1);
        if (transposed)
          out[((size_t)(b * HH + h) * DD + d) * LL + l] = (bf16)val;
        else
          out[((size_t)(b * HH + h) * LL + l) * DD + d] = (bf16)val;
      }
}

// ------------------------------- scores ------------------------------------
// z = h*B + b. A = qh[b,h] (L x D), BT = kh[b,h] (L rows of D). TEMP == 1.
__global__ __launch_bounds__(256) void k_gemm_scores(const bf16* qh, const bf16* kh,
                                                     const uint8_t* mask,
                                                     float* atten) {
  const int z = blockIdx.z;
  const int h = z >> 2, b = z & 3;
  const size_t hb = (size_t)(b * HH + h) * LL * DD;
  const int m0 = blockIdx.y * 128, n0 = blockIdx.x * 128;
  v8f acc[4][2] = {};
  gemm_core<false>(qh + hb, DD, kh + hb, DD, DD, m0, n0, acc);
  EPI_INDICES
#pragma unroll
  for (int i = 0; i < 4; ++i)
#pragma unroll
    for (int j = 0; j < 2; ++j)
#pragma unroll
      for (int e = 0; e < 8; ++e) {
        const int gm = m0 + wm * 64 + i * 16 + hl * 8 + e;
        const int gn = n0 + wn * 32 + j * 16 + r;
        float val = acc[i][j][e];
        if (mask[((size_t)b * LL + gm) * LL + gn]) val = -__builtin_inff();
        atten[((size_t)z * LL + gm) * LL + gn] = val;
      }
}

// ------------------------------- softmax -----------------------------------
__global__ __launch_bounds__(256) void k_softmax(float* atten) {
  float* p = atten + (size_t)blockIdx.x * LL;
  const int t = threadIdx.x;
  __shared__ float red[256];
  float vals[8];
  float mx = -__builtin_inff();
#pragma unroll
  for (int i = 0; i < 8; ++i) {
    vals[i] = p[t + i * 256];
    mx = fmaxf(mx, vals[i]);
  }
  red[t] = mx;
  __syncthreads();
  for (int s = 128; s > 0; s >>= 1) {
    if (t < s) red[t] = fmaxf(red[t], red[t + s]);
    __syncthreads();
  }
  mx = red[0];
  __syncthreads();
  float sum = 0.f;
#pragma unroll
  for (int i = 0; i < 8; ++i) {
    vals[i] = __expf(vals[i] - mx);
    sum += vals[i];
  }
  red[t] = sum;
  __syncthreads();
  for (int s = 128; s > 0; s >>= 1) {
    if (t < s) red[t] += red[t + s];
    __syncthreads();
  }
  const float inv = 1.0f / red[0];
#pragma unroll
  for (int i = 0; i < 8; ++i) p[t + i * 256] = vals[i] * inv;
}

// ------------------------------ ctx = atten·V ------------------------------
// A = atten[z] f32 (L x L), BT = vhT[b,h] (D rows of L). Out: ctx[BL, HD] bf16.
__global__ __launch_bounds__(256) void k_gemm_ctx(const float* atten,
                                                  const bf16* vhT, bf16* ctx) {
  const int z = blockIdx.z;
  const int h = z >> 2, b = z & 3;
  const int m0 = blockIdx.y * 128, n0 = blockIdx.x * 128;
  v8f acc[4][2] = {};
  gemm_core<true>(atten + (size_t)z * LL * LL, LL,
                  vhT + (size_t)(b * HH + h) * DD * LL, LL, LL, m0, n0, acc);
  EPI_INDICES
#pragma unroll
  for (int i = 0; i < 4; ++i)
#pragma unroll
    for (int j = 0; j < 2; ++j)
#pragma unroll
      for (int e = 0; e < 8; ++e) {
        const int gm = m0 + wm * 64 + i * 16 + hl * 8 + e;
        const int gn = n0 + wn * 32 + j * 16 + r;
        ctx[(size_t)(b * LL + gm) * HDc + h * DD + gn] = (bf16)acc[i][j][e];
      }
}

// ------------------------------ output proj --------------------------------
__global__ __launch_bounds__(256) void k_gemm_out(const bf16* ctx, const bf16* WoT,
                                                  const float* bo, float* proj) {
  const int m0 = blockIdx.y * 128, n0 = blockIdx.x * 128;
  v8f acc[4][2] = {};
  gemm_core<false>(ctx, HDc, WoT, HDc, HDc, m0, n0, acc);
  EPI_INDICES
#pragma unroll
  for (int i = 0; i < 4; ++i)
#pragma unroll
    for (int j = 0; j < 2; ++j)
#pragma unroll
      for (int e = 0; e < 8; ++e) {
        const int gm = m0 + wm * 64 + i * 16 + hl * 8 + e;
        const int gn = n0 + wn * 32 + j * 16 + r;
        proj[(size_t)gm * DD + gn] = acc[i][j][e] + bo[gn];
      }
}

// --------------------------- residual + layernorm --------------------------
__global__ __launch_bounds__(256) void k_ln(const float* proj, const float* resid,
                                            const float* gamma, const float* beta,
                                            float* out) {
  const int t = threadIdx.x;
  const size_t base = (size_t)blockIdx.x * DD;
  __shared__ float red[256];
  float x0 = proj[base + t] + resid[base + t];
  float x1 = proj[base + 256 + t] + resid[base + 256 + t];
  red[t] = x0 + x1;
  __syncthreads();
  for (int s = 128; s > 0; s >>= 1) {
    if (t < s) red[t] += red[t + s];
    __syncthreads();
  }
  const float mu = red[0] * (1.0f / DD);
  __syncthreads();
  const float d0 = x0 - mu, d1 = x1 - mu;
  red[t] = d0 * d0 + d1 * d1;
  __syncthreads();
  for (int s = 128; s > 0; s >>= 1) {
    if (t < s) red[t] += red[t + s];
    __syncthreads();
  }
  const float rstd = rsqrtf(red[0] * (1.0f / DD) + EPSc);
  out[base + t] = d0 * rstd * gamma[t] + beta[t];
  out[base + 256 + t] = d1 * rstd * gamma[256 + t] + beta[256 + t];
}

// ------------------------------- launcher ----------------------------------
extern "C" void kernel_launch(void* const* d_in, const int* in_sizes, int n_in,
                              void* d_out, int out_size, void* d_ws, size_t ws_size,
                              hipStream_t stream) {
  (void)in_sizes; (void)n_in; (void)out_size; (void)ws_size;
  const float*   q    = (const float*)d_in[0];
  const float*   k    = (const float*)d_in[1];
  const float*   v    = (const float*)d_in[2];
  const uint8_t* mask = (const uint8_t*)d_in[3];   // jnp bool -> 1 byte/elem
  const float*   Wq   = (const float*)d_in[4];
  const float*   bq   = (const float*)d_in[5];
  const float*   Wk   = (const float*)d_in[6];
  const float*   bk   = (const float*)d_in[7];
  const float*   Wv   = (const float*)d_in[8];
  const float*   bv   = (const float*)d_in[9];
  const float*   Wo   = (const float*)d_in[10];
  const float*   bo   = (const float*)d_in[11];
  const float*   gamma = (const float*)d_in[12];
  const float*   beta  = (const float*)d_in[13];

  // workspace layout (bytes)
  char* ws = (char*)d_ws;
  constexpr size_t SZ_XBF  = (size_t)BLc * CC * 2;          //  8 MB each
  constexpr size_t SZ_WT   = (size_t)CC * HDc * 2;          //  4 MB each
  constexpr size_t SZ_HEAD = (size_t)BB * HH * LL * DD * 2; // 64 MB each
  constexpr size_t SZ_CTX  = (size_t)BLc * HDc * 2;         // 64 MB
  constexpr size_t SZ_PROJ = (size_t)BLc * DD * 4;          // 16 MB
  size_t off = 0;
  bf16* qbf  = (bf16*)(ws + off); off += SZ_XBF;
  bf16* kbf  = (bf16*)(ws + off); off += SZ_XBF;
  bf16* vbf  = (bf16*)(ws + off); off += SZ_XBF;
  bf16* WqT  = (bf16*)(ws + off); off += SZ_WT;
  bf16* WkT  = (bf16*)(ws + off); off += SZ_WT;
  bf16* WvT  = (bf16*)(ws + off); off += SZ_WT;
  bf16* WoT  = (bf16*)(ws + off); off += SZ_WT;
  bf16* qh   = (bf16*)(ws + off); off += SZ_HEAD;
  bf16* kh   = (bf16*)(ws + off); off += SZ_HEAD;
  bf16* vhT  = (bf16*)(ws + off); off += SZ_HEAD;
  bf16* ctxb = (bf16*)(ws + off); off += SZ_CTX;
  float* proj = (float*)(ws + off); off += SZ_PROJ;

  float* outF  = (float*)d_out;                     // (B, L, D) f32
  float* atten = outF + (size_t)BB * LL * DD;       // (H*B, L, L) f32

  dim3 blk(256);

  // 0) converts
  const int nX = BLc * CC;
  k_cvt_bf16<<<(nX + 255) / 256, blk, 0, stream>>>(q, qbf, nX);
  k_cvt_bf16<<<(nX + 255) / 256, blk, 0, stream>>>(k, kbf, nX);
  k_cvt_bf16<<<(nX + 255) / 256, blk, 0, stream>>>(v, vbf, nX);
  const int nW = CC * HDc;
  k_transpose_cvt<<<(nW + 255) / 256, blk, 0, stream>>>(Wq, WqT, CC, HDc);
  k_transpose_cvt<<<(nW + 255) / 256, blk, 0, stream>>>(Wk, WkT, CC, HDc);
  k_transpose_cvt<<<(nW + 255) / 256, blk, 0, stream>>>(Wv, WvT, CC, HDc);
  k_transpose_cvt<<<(nW + 255) / 256, blk, 0, stream>>>(Wo, WoT, HDc, DD);

  // 1) QKV projections
  dim3 gproj(HDc / 128, BLc / 128);  // (32, 64)
  k_gemm_proj<<<gproj, blk, 0, stream>>>(qbf, WqT, bq, qh, 0);
  k_gemm_proj<<<gproj, blk, 0, stream>>>(kbf, WkT, bk, kh, 0);
  k_gemm_proj<<<gproj, blk, 0, stream>>>(vbf, WvT, bv, vhT, 1);

  // 2) scores + mask -> atten region of d_out
  dim3 gsc(LL / 128, LL / 128, HH * BB);  // (16, 16, 32)
  k_gemm_scores<<<gsc, blk, 0, stream>>>(qh, kh, mask, atten);

  // 3) softmax rows
  k_softmax<<<HH * BB * LL, blk, 0, stream>>>(atten);

  // 4) ctx = atten · V
  dim3 gctx(DD / 128, LL / 128, HH * BB);  // (4, 16, 32)
  k_gemm_ctx<<<gctx, blk, 0, stream>>>(atten, vhT, ctxb);

  // 5) output projection
  dim3 gout(DD / 128, BLc / 128);  // (4, 64)
  k_gemm_out<<<gout, blk, 0, stream>>>(ctxb, WoT, bo, proj);

  // 6) residual + layernorm -> d_out
  k_ln<<<BLc, blk, 0, stream>>>(proj, q, gamma, beta, outF);
}
